// SelfAttention_85169201480339
// MI455X (gfx1250) — matrix operand order repaired and checked by reading, not compiled, not run on hardware
//
#include <hip/hip_runtime.h>

// Problem constants (from reference): B=4, S=4096, DK=768, H=64.
#define BB  4
#define SS  4096
#define DKK 768
#define HH  64

typedef __attribute__((ext_vector_type(16))) __bf16    v16bf;
typedef __attribute__((ext_vector_type(8)))  unsigned  v8u;
typedef __attribute__((ext_vector_type(8)))  float     v8f;

// fp32 -> bf16 (RNE), scalar (epilogue stores only)
__device__ __forceinline__ unsigned short f2bf(float f) {
  unsigned u = __builtin_bit_cast(unsigned, f);
  u += 0x7FFFu + ((u >> 16) & 1u);
  return (unsigned short)(u >> 16);
}

// two fp32 -> packed bf16x2 (a -> [15:0], b -> [31:16]).
// Fallback: round-to-nearest (+0x8000) then one v_perm_b32 high-half merge.
__device__ __forceinline__ unsigned packbf2(float a, float b) {
#if __has_builtin(__builtin_amdgcn_cvt_pk_bf16_f32)
  auto r = __builtin_amdgcn_cvt_pk_bf16_f32(a, b);
  return __builtin_bit_cast(unsigned, r);
#else
  unsigned ua = __builtin_bit_cast(unsigned, a) + 0x8000u;
  unsigned ub = __builtin_bit_cast(unsigned, b) + 0x8000u;
  return __builtin_amdgcn_perm(ua, ub, 0x03020706u);
#endif
}

__device__ __forceinline__ v16bf asbf(v8u u) { return __builtin_bit_cast(v16bf, u); }

__device__ __forceinline__ v8f v8zero() {
  v8f z;
#pragma unroll
  for (int i = 0; i < 8; ++i) z[i] = 0.f;
  return z;
}

// B fragment (32x16, K = (j<8?j:j+8) + 8*half): dword d of lane = row32[(d<4?d:d+4) + 4*half]
__device__ __forceinline__ v16bf load_b16(const unsigned* __restrict__ row32, int half) {
  v8u u;
#pragma unroll
  for (int d = 0; d < 8; ++d) u[d] = row32[(d < 4 ? d : d + 4) + 4 * half];
  return asbf(u);
}

// A fragment (16x32, K = (j<8?j:j+8) + 8*half) from a bf16 row pointer (as u32)
__device__ __forceinline__ v16bf load_a16(const unsigned* __restrict__ row32, int half) {
  v8u u;
#pragma unroll
  for (int d = 0; d < 4; ++d) u[d] = row32[4 * half + d];
#pragma unroll
  for (int d = 0; d < 4; ++d) u[4 + d] = row32[8 + 4 * half + d];
  return asbf(u);
}

// ---------------------------------------------------------------------------
// Kernel 0: one-shot fp32 -> bf16 conversion of Wq and Wv.
// ---------------------------------------------------------------------------
__global__ __launch_bounds__(256)
void wcvt_kernel(const float* __restrict__ Wq, const float* __restrict__ Wv,
                 unsigned* __restrict__ Wqb, unsigned* __restrict__ Wvb) {
  const int n = HH * DKK / 2;
  int i = blockIdx.x * 256 + threadIdx.x;
  if (i < n) {
    Wqb[i] = packbf2(Wq[2 * i], Wq[2 * i + 1]);
    Wvb[i] = packbf2(Wv[2 * i], Wv[2 * i + 1]);
  }
}

// ---------------------------------------------------------------------------
// Kernel 1: fused q(=k) and v projections. [16384,768]x[768,64] bf16 WMMA,
// fp32 accumulate. Each wave -> one 16x64 output tile. q stored [B,S,H];
// v stored TRANSPOSED [B,H,S] so the attention P.V B-fragments are contiguous.
// ---------------------------------------------------------------------------
__global__ __launch_bounds__(128)
void proj_kernel(const float* __restrict__ x,
                 const unsigned* __restrict__ Wqb, const unsigned* __restrict__ Wvb,
                 const float* __restrict__ bq, const float* __restrict__ bv,
                 unsigned short* __restrict__ qk, unsigned short* __restrict__ vt) {
  const int lane = threadIdx.x & 31;
  const int wave = threadIdx.x >> 5;
  const int half = lane >> 4;
  const int lm   = lane & 15;
  const int row0 = blockIdx.x * 64 + wave * 16;
  const float* xr = x + (size_t)(row0 + lm) * DKK;

  v8f accQ[4], accV[4];
#pragma unroll
  for (int i = 0; i < 4; ++i) { accQ[i] = v8zero(); accV[i] = v8zero(); }

  for (int dc = 0; dc < DKK; dc += 32) {
    v16bf fq[4], fv[4];
#pragma unroll
    for (int hc = 0; hc < 4; ++hc) {
      int h = hc * 16 + lm;
      fq[hc] = load_b16(Wqb + (size_t)h * (DKK / 2) + dc / 2, half);
      fv[hc] = load_b16(Wvb + (size_t)h * (DKK / 2) + dc / 2, half);
    }
    float4 a0 = *(const float4*)(xr + dc + 8 * half);
    float4 a1 = *(const float4*)(xr + dc + 8 * half + 4);
    float4 a2 = *(const float4*)(xr + dc + 16 + 8 * half);
    float4 a3 = *(const float4*)(xr + dc + 16 + 8 * half + 4);
    v8u ua;
    ua[0] = packbf2(a0.x, a0.y); ua[1] = packbf2(a0.z, a0.w);
    ua[2] = packbf2(a1.x, a1.y); ua[3] = packbf2(a1.z, a1.w);
    ua[4] = packbf2(a2.x, a2.y); ua[5] = packbf2(a2.z, a2.w);
    ua[6] = packbf2(a3.x, a3.y); ua[7] = packbf2(a3.z, a3.w);
    v16bf va = asbf(ua);

#pragma unroll
    for (int hc = 0; hc < 4; ++hc) {
      accQ[hc] = __builtin_amdgcn_wmma_f32_16x16x32_bf16(
          false, va, false, fq[hc], (short)0, accQ[hc], false, false);
      accV[hc] = __builtin_amdgcn_wmma_f32_16x16x32_bf16(
          false, va, false, fv[hc], (short)0, accV[hc], false, false);
    }
  }

#pragma unroll
  for (int hc = 0; hc < 4; ++hc) {
    int h = hc * 16 + lm;
    float bqv = bq[h], bvv = bv[h];
#pragma unroll
    for (int g = 0; g < 8; ++g) {
      int row = row0 + g + 8 * half;
      int bb  = row >> 12;
      int r   = row & (SS - 1);
      qk[(size_t)row * HH + h] = f2bf(accQ[hc][g] + bqv);
      vt[(size_t)bb * HH * SS + (size_t)h * SS + r] = f2bf(accV[hc][g] + bvv);
    }
  }
}

// ---------------------------------------------------------------------------
// Kernel 2: flash attention with 64-key tiles. score = q.k^T * 768^-0.5,
// online softmax in exp2 domain, out = p.v.  k == q (source bug preserved).
// Row sums of P computed by WMMA against a ones matrix (no shuffle chain).
// 4 waves/WG share async-loaded K and V^T LDS tiles; each wave = 16 queries.
// ---------------------------------------------------------------------------
__global__ __launch_bounds__(128)
void attn_kernel(const unsigned short* __restrict__ qk,
                 const unsigned short* __restrict__ vt,
                 float* __restrict__ out) {
  __shared__ __align__(16) unsigned short sK[64 * 64];   // 8 KB: [key][h]
  __shared__ __align__(16) unsigned short sVT[64 * 64];  // 8 KB: [h][key]
  __shared__ float sP[4][16 * 64];                       // 16 KB P stage

  const int lane = threadIdx.x & 31;
  const int wave = threadIdx.x >> 5;
  const int half = lane >> 4;
  const int lm   = lane & 15;
  const int b    = blockIdx.x >> 6;
  const int qt   = blockIdx.x & 63;
  const int q0   = qt * 64 + wave * 16;
  const size_t base = (size_t)b * SS * HH;

  // Q in bf16 A-layout; K-dim = head dim (64) -> two 16x32 A tiles.
  const unsigned* rowq = (const unsigned*)qk + (base + (size_t)(q0 + lm) * HH) / 2;
  v16bf aq0 = load_a16(rowq, half);
  v16bf aq1 = load_a16(rowq + 16, half);

  // ones(32x16) bf16 B fragment for row-sum WMMA
  v8u uo;
#pragma unroll
  for (int d = 0; d < 8; ++d) uo[d] = 0x3F803F80u;
  const v16bf ones = asbf(uo);

  float mi[8], li[8];
  v8f o[4];
#pragma unroll
  for (int g = 0; g < 8; ++g) { mi[g] = -__builtin_inff(); li[g] = 0.f; }
#pragma unroll
  for (int i = 0; i < 4; ++i) o[i] = v8zero();

  // 768^-0.5 folded with log2(e): softmax in exp2 domain.
  const float scl2 = 0.036084391824351614f * 1.4426950408889634f;
  const unsigned ldsK  = (unsigned)(size_t)&sK[0];
  const unsigned ldsVT = (unsigned)(size_t)&sVT[0];
  float* pst = &sP[wave][0];

  for (int kt = 0; kt < SS; kt += 64) {
    __syncthreads();  // previous tile consumers done
    {
      // Async copy: K tile (4096 contiguous bf16 = 512x16B) and V^T tile
      // (64 h-rows x 64 keys = 8 x 16B per row) -> LDS, tracked by ASYNCcnt.
      const unsigned short* kg  = qk + base + (size_t)kt * HH;
      const unsigned short* vtg = vt + (size_t)b * HH * SS + kt;
      for (int c = threadIdx.x; c < 512; c += 128) {
        unsigned long long ga = (unsigned long long)(kg + c * 8);
        unsigned la = ldsK + c * 16;
        asm volatile("global_load_async_to_lds_b128 %0, %1, off"
                     :: "v"(la), "v"(ga) : "memory");
        ga = (unsigned long long)(vtg + (size_t)(c >> 3) * SS + (c & 7) * 8);
        la = ldsVT + c * 16;
        asm volatile("global_load_async_to_lds_b128 %0, %1, off"
                     :: "v"(la), "v"(ga) : "memory");
      }
      if (kt + 64 < SS) __builtin_prefetch(kg + 64 * HH, 0, 1);
      asm volatile("s_wait_asynccnt 0" ::: "memory");
    }
    __syncthreads();  // tiles visible to all waves

    // Scores: 16x64 = Q(16x64) . K^T(64x64): 4 C tiles x 2 WMMA each.
    const unsigned* sk32 = (const unsigned*)sK;
    v8f cs[4];
#pragma unroll
    for (int n = 0; n < 4; ++n) {
      const unsigned* rn = sk32 + (size_t)(n * 16 + lm) * 32;  // key column rows
      v16bf f0 = load_b16(rn, half);       // h 0..31
      v16bf f1 = load_b16(rn + 16, half);  // h 32..63
      v8f c = v8zero();
      c = __builtin_amdgcn_wmma_f32_16x16x32_bf16(false, aq0, false, f0, (short)0, c, false, false);
      c = __builtin_amdgcn_wmma_f32_16x16x32_bf16(false, aq1, false, f1, (short)0, c, false, false);
      cs[n] = c;
    }

    // Online softmax (exp2 domain).  Row max via 16-lane shfl_xor; exp'd
    // probabilities written straight into the per-wave P stage.
#pragma unroll
    for (int g = 0; g < 8; ++g) {
      float s0 = cs[0][g] * scl2, s1 = cs[1][g] * scl2;
      float s2 = cs[2][g] * scl2, s3 = cs[3][g] * scl2;
      float rm = fmaxf(fmaxf(s0, s1), fmaxf(s2, s3));
#pragma unroll
      for (int m = 1; m < 16; m <<= 1) rm = fmaxf(rm, __shfl_xor(rm, m, 32));
      float nm = fmaxf(mi[g], rm);
      float corr = exp2f(mi[g] - nm);
      mi[g] = nm;
      li[g] *= corr;
#pragma unroll
      for (int t = 0; t < 4; ++t) o[t][g] *= corr;
      int row = g + 8 * half;
      pst[row * 64 + lm]      = exp2f(s0 - nm);
      pst[row * 64 + 16 + lm] = exp2f(s1 - nm);
      pst[row * 64 + 32 + lm] = exp2f(s2 - nm);
      pst[row * 64 + 48 + lm] = exp2f(s3 - nm);
    }
    asm volatile("s_wait_dscnt 0" ::: "memory");  // wave-local LDS RAW ordering

    // Re-marshal P -> two bf16 A fragments (keys 0..31, 32..63).
    v8u up0, up1;
#pragma unroll
    for (int d = 0; d < 8; ++d) {
      int k = (d < 4 ? 2 * d : 2 * d + 8) + 8 * half;
      up0[d] = packbf2(pst[lm * 64 + k],      pst[lm * 64 + k + 1]);
      up1[d] = packbf2(pst[lm * 64 + 32 + k], pst[lm * 64 + 32 + k + 1]);
    }
    v16bf ap0 = asbf(up0), ap1 = asbf(up1);

    // Row sums of (bf16) P via WMMA against ones -> every column = row sum.
    v8f csum = v8zero();
    csum = __builtin_amdgcn_wmma_f32_16x16x32_bf16(false, ap0, false, ones, (short)0, csum, false, false);
    csum = __builtin_amdgcn_wmma_f32_16x16x32_bf16(false, ap1, false, ones, (short)0, csum, false, false);
#pragma unroll
    for (int g = 0; g < 8; ++g) li[g] += csum[g];

    // O += P(16x64) . V(64x64): V^T rows give contiguous B fragments.
    const unsigned* sv32 = (const unsigned*)sVT;
#pragma unroll
    for (int hc = 0; hc < 4; ++hc) {
      const unsigned* rv = sv32 + (size_t)(hc * 16 + lm) * 32;  // 64 keys/row
      v16bf g0 = load_b16(rv, half);       // keys 0..31
      v16bf g1 = load_b16(rv + 16, half);  // keys 32..63
      o[hc] = __builtin_amdgcn_wmma_f32_16x16x32_bf16(false, ap0, false, g0, (short)0, o[hc], false, false);
      o[hc] = __builtin_amdgcn_wmma_f32_16x16x32_bf16(false, ap1, false, g1, (short)0, o[hc], false, false);
    }
  }

  // Normalize (one reciprocal per row) and store fp32 output [B,S,H].
#pragma unroll
  for (int g = 0; g < 8; ++g) {
    float inv = 1.0f / li[g];
    int row = q0 + g + 8 * half;
#pragma unroll
    for (int hc = 0; hc < 4; ++hc)
      out[base + (size_t)row * HH + hc * 16 + lm] = o[hc][g] * inv;
  }
}

// ---------------------------------------------------------------------------
extern "C" void kernel_launch(void* const* d_in, const int* in_sizes, int n_in,
                              void* d_out, int out_size, void* d_ws, size_t ws_size,
                              hipStream_t stream) {
  (void)in_sizes; (void)n_in; (void)out_size; (void)ws_size;
  const float* x  = (const float*)d_in[0];
  const float* Wq = (const float*)d_in[1];
  const float* bq = (const float*)d_in[2];
  const float* Wv = (const float*)d_in[3];
  const float* bv = (const float*)d_in[4];

  // Workspace: qk bf16 [B*S,64] (2 MB) | v^T bf16 [B,64,S] (2 MB)
  //          | Wq bf16 (96 KB) | Wv bf16 (96 KB)
  unsigned short* qkb = (unsigned short*)d_ws;
  unsigned short* vtb = qkb + (size_t)BB * SS * HH;
  unsigned*       Wqb = (unsigned*)(vtb + (size_t)BB * SS * HH);
  unsigned*       Wvb = Wqb + (size_t)HH * DKK / 2;

  wcvt_kernel<<<dim3((HH * DKK / 2 + 255) / 256), dim3(256), 0, stream>>>(Wq, Wv, Wqb, Wvb);
  proj_kernel<<<dim3((BB * SS) / 64), dim3(128), 0, stream>>>(x, Wqb, Wvb, bq, bv, qkb, vtb);
  attn_kernel<<<dim3(BB * (SS / 64)), dim3(128), 0, stream>>>(qkb, vtb, (float*)d_out);
}
